// TransformerModelV11SkipLast_35742717837439
// MI455X (gfx1250) — matrix operand order repaired and checked by reading, not compiled, not run on hardware
//
#include <hip/hip_runtime.h>
#include <hip/hip_bf16.h>

// Problem constants (from reference)
#define BB 16
#define DD 512
#define TT 2048
#define HH 2
#define HD 256   // DD/HH

typedef __bf16 bf16_t;
typedef __attribute__((ext_vector_type(16))) __bf16 v16bf;
typedef __attribute__((ext_vector_type(8)))  float  v8f;

// LDS B-tile: 16 rows (N) x 512 K, rows padded to 520 bf16 so that the
// 16 half-wave lanes (stride 260 dwords, 260%64==4 banks) hit distinct banks.
#define BROW 520

// ---------------------------------------------------------------------------
// WMMA fragment loaders per CDNA5 ISA layouts (CDNA5 ISA)
// A (16x32 bf16, M x K): lanes 0-15 = rows M=0..15 holding K = k0+{0..7,16..23},
//                        lanes 16-31 = same rows holding K = k0+{8..15,24..31}
// B (32x16 bf16, K x N): lane L<16 = col N=L holding K = k0+{0..7,16..23},
//                        lane L>=16 = col N=L-16 holding K = k0+{8..15,24..31}
// ---------------------------------------------------------------------------
__device__ __forceinline__ v16bf load_frag_A(const bf16_t* base, int lda,
                                             int row0, int k0, int lane) {
    const int  lo = lane & 15;
    const bool hi = lane >= 16;
    const bf16_t* p = base + (size_t)(row0 + lo) * lda + k0 + (hi ? 8 : 0);
    v16bf r;
    ((uint4*)&r)[0] = *(const uint4*)(p);        // K chunk k0+off .. +8
    ((uint4*)&r)[1] = *(const uint4*)(p + 16);   // K chunk k0+16+off .. +8
    return r;
}

// B fragment from the LDS-staged tile (row = N-col within tile, contiguous K)
__device__ __forceinline__ v16bf frag_B_lds(const unsigned short* Bs, int k0,
                                            int lane) {
    const int  lo = lane & 15;
    const bool hi = lane >= 16;
    const unsigned short* p = Bs + lo * BROW + k0 + (hi ? 8 : 0);
    v16bf r;
    ((uint4*)&r)[0] = *(const uint4*)(p);        // K chunk k0+off .. +8
    ((uint4*)&r)[1] = *(const uint4*)(p + 16);   // K chunk k0+16+off .. +8
    return r;
}

// Cooperative staging of a 16x512 bf16 tile (16KB) through registers.
// Thread tid owns rows (tid>>6)+2i at fixed column (tid&63)*8, so all 8
// global loads / LDS stores share ONE base address + immediate offsets
// (stride 2*DD elements / 2*BROW elements) -> batched s_clause loads.
__device__ __forceinline__ void stage_load(const bf16_t* rowbase, uint4* stage,
                                           int tid) {
    const uint4* p =
        (const uint4*)(rowbase + (size_t)(tid >> 6) * DD + ((tid & 63) << 3));
#pragma unroll
    for (int i = 0; i < 8; ++i) stage[i] = p[i * (2 * DD / 8)];
}

__device__ __forceinline__ void stage_store(unsigned short* Bs,
                                            const uint4* stage, int tid) {
    uint4* p = (uint4*)(Bs + (tid >> 6) * BROW + ((tid & 63) << 3));
#pragma unroll
    for (int i = 0; i < 8; ++i) p[i * (2 * BROW / 8)] = stage[i];
}

// ---------------------------------------------------------------------------
// Kernel: tiled transpose fp32 -> bf16.  dst[c*R + r] = bf16(src[r*C + c])
// grid: (C/32, R/32, batches), block: (32, 8)
// ---------------------------------------------------------------------------
__global__ void k_transpose_bf16(const float* __restrict__ src,
                                 bf16_t* __restrict__ dst, int R, int C) {
    __shared__ float tile[32][33];
    const size_t bofs = (size_t)blockIdx.z * R * C;
    src += bofs; dst += bofs;
    const int c0 = blockIdx.x * 32, r0 = blockIdx.y * 32;
    const int tx = threadIdx.x, ty = threadIdx.y;
#pragma unroll
    for (int i = 0; i < 4; ++i)
        tile[ty + i * 8][tx] = src[(size_t)(r0 + ty + i * 8) * C + (c0 + tx)];
    __syncthreads();
#pragma unroll
    for (int i = 0; i < 4; ++i)
        dst[(size_t)(c0 + ty + i * 8) * R + (r0 + tx)] =
            (bf16_t)tile[tx][ty + i * 8];
}

__global__ void k_zero(float* __restrict__ p, int n) {
    int i = blockIdx.x * blockDim.x + threadIdx.x;
    if (i < n) p[i] = 0.0f;
}

// ---------------------------------------------------------------------------
// Kernel: Q[b] (2048x512 bf16) = Xt[b] (2048x512 bf16) @ W_KQ (512x512)
// W_KQ pre-transposed to N-major (WKQt[j][d]).
// grid: (T/64, B), block: 128 (4 waves). Each wave owns one 16-row M-tile;
// the block loops all 32 N-tiles, staging each 16x512 B tile in LDS once.
// Two independent accumulation chains (even/odd k) overlap ds latency.
// ---------------------------------------------------------------------------
__global__ void __launch_bounds__(128, 1)
k_gemm_q(const bf16_t* __restrict__ Xt,
         const bf16_t* __restrict__ WKQt,
         bf16_t* __restrict__ Q) {
    __shared__ __align__(16) unsigned short Bs[16 * BROW];
    const int tid  = threadIdx.x;
    const int lane = tid & 31;
    const int wave = tid >> 5;
    const int lo   = lane & 15;
    const bool hiL = lane >= 16;
    const int s0   = blockIdx.x * 64 + wave * 16;
    const size_t bofs = (size_t)blockIdx.y * TT * DD;
    const bf16_t* A = Xt + bofs;
    bf16_t*       O = Q  + bofs;

    // Resident A fragments over full K=512
    v16bf a[16];
#pragma unroll
    for (int k = 0; k < 16; ++k) a[k] = load_frag_A(A, DD, s0, k * 32, lane);

    uint4 stage[8];
    stage_load(WKQt, stage, tid);               // N-tile 0
    for (int nt = 0; nt < 32; ++nt) {
        __syncthreads();                        // previous tile fully consumed
        stage_store(Bs, stage, tid);
        __syncthreads();
        if (nt + 1 < 32)                        // prefetch next tile into regs
            stage_load(WKQt + (size_t)(nt + 1) * 16 * DD, stage, tid);

        v8f ce = {}, co = {};                   // two independent chains
#pragma unroll
        for (int k = 0; k < 8; ++k) {
            v16bf b0 = frag_B_lds(Bs, (2 * k) * 32, lane);
            ce = __builtin_amdgcn_wmma_f32_16x16x32_bf16(false, a[2 * k], false, b0,
                                                         (short)0, ce, false, false);
            v16bf b1 = frag_B_lds(Bs, (2 * k + 1) * 32, lane);
            co = __builtin_amdgcn_wmma_f32_16x16x32_bf16(false, a[2 * k + 1], false, b1,
                                                         (short)0, co, false, false);
        }
        const int n0 = nt * 16;
#pragma unroll
        for (int v = 0; v < 8; ++v) {
            const int m = v + (hiL ? 8 : 0);
            O[(size_t)(s0 + m) * DD + n0 + lo] = (bf16_t)(ce[v] + co[v]);
        }
    }
}

// ---------------------------------------------------------------------------
// Kernel: fused attention-score + 2-head softmax + contraction with w.
//   attn_h[s,t] = Q[b][s, h*256 .. ] . Xt[b][t, h*256 .. ]   (K = 256)
//   sm0 = 1/(1+exp(a1-a0)), sm1 = 1-sm0
//   g[b][h][s] += sum_t sm_h * w[t]        (mask s == T-1)
// grid: (T/128, T/64, B), block: 128 (4 waves).
// 4 waves = 4 consecutive s-tiles sharing one 128-column t-chunk; each
// 16x512 Xt B tile (both heads' K) is staged in LDS once and reused 4x.
// The c0/c1 head chains are independent -> ds latency overlaps WMMA.
// ---------------------------------------------------------------------------
__global__ void __launch_bounds__(128, 1)
k_attn_g(const bf16_t* __restrict__ Q,
         const bf16_t* __restrict__ Xt,
         const float* __restrict__ wlast,
         float* __restrict__ g) {
    __shared__ __align__(16) unsigned short Bs[16 * BROW];
    const int tid  = threadIdx.x;
    const int lane = tid & 31;
    const int wave = tid >> 5;
    const int lo   = lane & 15;
    const bool hiL = lane >= 16;
    const int s0   = blockIdx.y * 64 + wave * 16;
    const int tch  = blockIdx.x * 128;
    const int b    = blockIdx.z;
    const size_t bofs = (size_t)b * TT * DD;
    const bf16_t* Qb = Q + bofs;
    const bf16_t* Xb = Xt + bofs;
    float* g0 = g + (size_t)(b * 2 + 0) * TT;
    float* g1 = g + (size_t)(b * 2 + 1) * TT;

    // Resident A fragments for both heads over full K=256
    v16bf a0[8], a1[8];
#pragma unroll
    for (int k = 0; k < 8; ++k) {
        a0[k] = load_frag_A(Qb, DD, s0, 0   + k * 32, lane);
        a1[k] = load_frag_A(Qb, DD, s0, 256 + k * 32, lane);
    }

    float gacc0[8], gacc1[8];
#pragma unroll
    for (int v = 0; v < 8; ++v) { gacc0[v] = 0.0f; gacc1[v] = 0.0f; }

    uint4 stage[8];
    stage_load(Xb + (size_t)tch * DD, stage, tid);      // t-tile 0
    for (int nt = 0; nt < 8; ++nt) {
        __syncthreads();
        stage_store(Bs, stage, tid);
        __syncthreads();
        if (nt + 1 < 8)
            stage_load(Xb + (size_t)(tch + (nt + 1) * 16) * DD, stage, tid);

        v8f c0 = {}, c1 = {};
#pragma unroll
        for (int k = 0; k < 8; ++k) {
            v16bf b0 = frag_B_lds(Bs, 0   + k * 32, lane);
            c0 = __builtin_amdgcn_wmma_f32_16x16x32_bf16(false, a0[k], false, b0,
                                                         (short)0, c0, false, false);
            v16bf b1 = frag_B_lds(Bs, 256 + k * 32, lane);
            c1 = __builtin_amdgcn_wmma_f32_16x16x32_bf16(false, a1[k], false, b1,
                                                         (short)0, c1, false, false);
        }
        const float wv = wlast[tch + nt * 16 + lo];     // lane's N column weight
#pragma unroll
        for (int v = 0; v < 8; ++v) {
            const float d   = c1[v] - c0[v];
            const float e   = __expf(d);
            const float sm0 = 1.0f / (1.0f + e);        // stable 2-way softmax
            const float sm1 = 1.0f - sm0;
            gacc0[v] += sm0 * wv;
            gacc1[v] += sm1 * wv;
        }
    }

    // Reduce over the 16 N-lanes of each half-wave, then fold into g.
#pragma unroll
    for (int v = 0; v < 8; ++v) {
        float r0 = gacc0[v], r1 = gacc1[v];
#pragma unroll
        for (int m = 1; m < 16; m <<= 1) {
            r0 += __shfl_xor(r0, m, 32);
            r1 += __shfl_xor(r1, m, 32);
        }
        if (lo == 0) {
            const int s = s0 + v + (hiL ? 8 : 0);
            if (s < TT - 1) {           // xs_skip: exclude s = T-1
                atomicAdd(&g0[s], r0);
                atomicAdd(&g1[s], r1);
            }
        }
    }
}

// ---------------------------------------------------------------------------
// Kernel: y[b][h][k] = sum_{s<T-1} xs[b][k][s] * g[b][h][s]
// grid: (D, B), block: 256
// ---------------------------------------------------------------------------
__global__ void k_y(const float* __restrict__ xs, const float* __restrict__ g,
                    float* __restrict__ y) {
    __shared__ float r0s[256], r1s[256];
    const int k = blockIdx.x, b = blockIdx.y, tid = threadIdx.x;
    const float* xrow = xs + ((size_t)b * DD + k) * TT;
    const float* g0 = g + (size_t)(b * 2 + 0) * TT;
    const float* g1 = g + (size_t)(b * 2 + 1) * TT;
    float a0 = 0.0f, a1 = 0.0f;
    for (int s = tid; s < TT - 1; s += 256) {
        const float x = xrow[s];
        a0 += x * g0[s];
        a1 += x * g1[s];
    }
    r0s[tid] = a0; r1s[tid] = a1;
    __syncthreads();
    for (int off = 128; off > 0; off >>= 1) {
        if (tid < off) { r0s[tid] += r0s[tid + off]; r1s[tid] += r1s[tid + off]; }
        __syncthreads();
    }
    if (tid == 0) {
        y[(size_t)(b * 2 + 0) * DD + k] = r0s[0];
        y[(size_t)(b * 2 + 1) * DD + k] = r1s[0];
    }
}

// ---------------------------------------------------------------------------
// Kernel: out[b][d] = sum_k W_PV[d][k] * y[b][d>>8][k]
// grid: B*D blocks of 32 (one wave each)
// ---------------------------------------------------------------------------
__global__ void k_out(const float* __restrict__ WPV, const float* __restrict__ y,
                      float* __restrict__ out) {
    const int id = blockIdx.x;
    const int b = id >> 9, d = id & (DD - 1);
    const int h = d >> 8;
    const int lane = threadIdx.x;
    const float* wrow = WPV + (size_t)d * DD;
    const float* yrow = y + (size_t)(b * 2 + h) * DD;
    float acc = 0.0f;
#pragma unroll
    for (int k = lane; k < DD; k += 32) acc += wrow[k] * yrow[k];
#pragma unroll
    for (int m = 1; m < 32; m <<= 1) acc += __shfl_xor(acc, m, 32);
    if (lane == 0) out[(size_t)b * DD + d] = acc;
}

// ---------------------------------------------------------------------------
extern "C" void kernel_launch(void* const* d_in, const int* in_sizes, int n_in,
                              void* d_out, int out_size, void* d_ws, size_t ws_size,
                              hipStream_t stream) {
    const float* xs   = (const float*)d_in[0];   // (B, D, T)
    const float* W_KQ = (const float*)d_in[1];   // (D, D)
    const float* W_PV = (const float*)d_in[2];   // (D, D)
    const float* W_out = (const float*)d_in[3];  // (T, T); only last row used
    float* out = (float*)d_out;                  // (B, D)

    // Workspace carve-up
    char* ws = (char*)d_ws;
    const size_t XT_BYTES = (size_t)BB * TT * DD * sizeof(bf16_t); // 32 MB
    const size_t Q_BYTES  = XT_BYTES;                              // 32 MB
    const size_t WK_BYTES = (size_t)DD * DD * sizeof(bf16_t);      // 512 KB
    const size_t G_BYTES  = (size_t)BB * HH * TT * sizeof(float);  // 256 KB
    bf16_t* Xt   = (bf16_t*)(ws);
    bf16_t* Q    = (bf16_t*)(ws + XT_BYTES);
    bf16_t* WKQt = (bf16_t*)(ws + XT_BYTES + Q_BYTES);
    float*  g    = (float*) (ws + XT_BYTES + Q_BYTES + WK_BYTES);
    float*  y    = (float*) (ws + XT_BYTES + Q_BYTES + WK_BYTES + G_BYTES);

    const float* wlast = W_out + (size_t)(TT - 1) * TT;

    // 1) Pack: Xt[b][t][d] = bf16(xs[b][d][t]) ; WKQt[j][d] = bf16(W_KQ[d][j])
    k_transpose_bf16<<<dim3(TT / 32, DD / 32, BB), dim3(32, 8), 0, stream>>>(
        xs, Xt, DD, TT);
    k_transpose_bf16<<<dim3(DD / 32, DD / 32, 1), dim3(32, 8), 0, stream>>>(
        W_KQ, WKQt, DD, DD);

    // 2) g := 0 (atomically accumulated below; must be re-zeroed every call)
    k_zero<<<(BB * HH * TT + 255) / 256, 256, 0, stream>>>(g, BB * HH * TT);

    // 3) Q = Xt @ W_KQ  (bf16 WMMA, f32 accumulate, LDS-staged B)
    k_gemm_q<<<dim3(TT / 64, BB, 1), 128, 0, stream>>>(Xt, WKQt, Q);

    // 4) Fused attn scores + 2-head softmax + g contraction (dominant GEMM)
    k_attn_g<<<dim3(TT / 128, TT / 64, BB), 128, 0, stream>>>(Q, Xt, wlast, g);

    // 5) y[b][h][k] = sum_s xs[b][k][s] * g[b][h][s]
    k_y<<<dim3(DD, BB, 1), 256, 0, stream>>>(xs, g, y);

    // 6) out = W_PV . y
    k_out<<<BB * DD, 32, 0, stream>>>(W_PV, y, out);
}
